// GCN_spatial_subnet_61916248539650
// MI455X (gfx1250) — matrix operand order repaired and checked by reading, not compiled, not run on hardware
//
#include <hip/hip_runtime.h>
#include <math.h>

// ---------------------------------------------------------------------------
// CDNA5 (gfx1250, wave32) implementation of the 10-layer GCN spatial subnet.
// Channel GEMMs + attention Gram reduction use v_wmma_f32_16x16x32_bf16.
// LDS tiles are stored in WMMA *fragment order* so each lane reads its whole
// 16-element bf16 fragment as one contiguous 32B load (2x ds_load_b128).
// The GEMM B-tile (strided 2D tile of X) is fetched by the Tensor Data Mover
// (tensor_load_to_lds + s_wait_tensorcnt) when the builtin is available.
// ---------------------------------------------------------------------------

typedef __bf16 bf16_t;
typedef __attribute__((ext_vector_type(16))) __bf16 v16bf;
typedef __attribute__((ext_vector_type(4)))  __bf16 v4bf;
typedef __attribute__((ext_vector_type(8)))  float  v8f;

#define HAVE_TDM __has_builtin(__builtin_amdgcn_tensor_load_to_lds)
#if HAVE_TDM
typedef __attribute__((ext_vector_type(4))) unsigned int v4u;
typedef __attribute__((ext_vector_type(8))) int          v8i_t;
typedef __attribute__((ext_vector_type(4))) int          v4i_t;
#endif

// Fragment-order index helpers (inverse of CDNA5 ISA 7.12.2 layouts).
//   A 16x32 (m,k):  lane = (m&15) + 16*k[3],  slot = 8*k[4] + (k&7)
//   B 32x16 (k,n):  lane = (n&15) + 16*k[3],  slot = 8*k[4] + (k&7)
// Storage per 16-wide subtile: 32 lanes x 16 slots = 512 bf16 (1 KB).
__device__ __forceinline__ int fragA_idx(int m, int k) {
  return ((m >> 4) * 512) + (((m & 15) + 16 * ((k >> 3) & 1)) * 16)
       + (((k >> 4) & 1) * 8) + (k & 7);
}
__device__ __forceinline__ int fragB_idx(int k, int n) {
  return ((n >> 4) * 512) + (((n & 15) + 16 * ((k >> 3) & 1)) * 16)
       + (((k >> 4) & 1) * 8) + (k & 7);
}

#define WMMA_BF16(A, B, C) \
  __builtin_amdgcn_wmma_f32_16x16x32_bf16(false, (A), false, (B), (short)0, (C), false, false)

// ---------------------------------------------------------------------------
// GEMM: Y[n,o,s] = sum_c W[o,c] * X[n,c,s] + bias[o] (+= Y if ACC)
// grid (S/64, O/64, N), block 128 (4 waves, each a 32x32 accumulator)
// ---------------------------------------------------------------------------
template <int ACC>
__global__ __launch_bounds__(128)
void gemm_proj_kernel(float* __restrict__ Y, const float* __restrict__ W,
                      const float* __restrict__ bias, const float* __restrict__ X,
                      int O, int C, int S)
{
  __shared__ __align__(32) bf16_t Ws[4 * 512];   // A tile 64x32, fragment order
  __shared__ __align__(32) bf16_t Xs[4 * 512];   // B tile 32x64, fragment order
#if HAVE_TDM
  __shared__ __align__(16) float  Xf[32 * 64];   // raw f32 B tile (TDM destination)
#endif
  const int t = threadIdx.x;
  const int lane = t & 31;
  const int wv = t >> 5;
  const int n  = blockIdx.z;
  const int s0 = blockIdx.x * 64;
  const int o0 = blockIdx.y * 64;
  const float* Xn = X + (size_t)n * C * S;
  float*       Yn = Y + (size_t)n * O * S;
  const int wm = (wv >> 1) * 32;   // wave M offset in 64-tile
  const int wn = (wv & 1) * 32;    // wave N offset

  v8f acc[2][2] = {};

  for (int c0 = 0; c0 < C; c0 += 32) {
    // --- W tile 64(o) x 32(k): float4 load, packed v4bf store (slots contig) ---
#pragma unroll
    for (int r = 0; r < 4; ++r) {
      int idx4 = t + r * 128;             // 0..511 float4s
      int ol = idx4 >> 3;                 // 0..63
      int k4 = (idx4 & 7) * 4;            // 0,4,..,28 (4 consecutive k, same lane)
      const float4 w4 = *(const float4*)(W + (size_t)(o0 + ol) * C + c0 + k4);
      v4bf v = {(bf16_t)w4.x, (bf16_t)w4.y, (bf16_t)w4.z, (bf16_t)w4.w};
      *(v4bf*)(Ws + fragA_idx(ol, k4)) = v;
    }

#if HAVE_TDM
    // --- X tile 32(k) x 64(s) via Tensor Data Mover: one DMA per tile -------
    // D# per CDNA5 ISA sec.8: 2D tensor [C rows x S cols],
    // tile 32x64 f32 at (c0, s0), contiguous 8KB LDS destination.
    if (t < 32) {                          // single wave issues the DMA
      const unsigned long long ga =
          (unsigned long long)(size_t)(Xn + (size_t)c0 * S + s0);
      const unsigned lds = (unsigned)(size_t)(void*)Xf;
      v4u g0;
      g0[0] = 1u;                          // count=1 valid descriptor
      g0[1] = lds;                         // lds_addr  (bits 63:32)
      g0[2] = (unsigned)(ga & 0xFFFFFFFFu);               // global_addr lo
      g0[3] = (unsigned)((ga >> 32) & 0x01FFFFFFu)        // global_addr[56:32]
            | (2u << 30);                                 // type = 2 (image)
      v8i_t g1;
      g1[0] = (int)(2u << 16);             // data_size = 2 -> 4 bytes
      g1[1] = (int)(((unsigned)S & 0xFFFFu) << 16);       // tensor_dim0[15:0]
      g1[2] = (int)((((unsigned)S >> 16) & 0xFFFFu)       // tensor_dim0[31:16]
            | (((unsigned)C & 0xFFFFu) << 16));           // tensor_dim1[15:0]
      g1[3] = (int)((((unsigned)C >> 16) & 0xFFFFu)       // tensor_dim1[31:16]
            | (64u << 16));                               // tile_dim0 = 64
      g1[4] = (int)32u;                    // tile_dim1 = 32, tile_dim2 = 0
      g1[5] = (int)S;                      // tensor_dim0_stride[31:0] = S
      g1[6] = 0;                           // stride[47:32] | dim1_stride lo
      g1[7] = 0;
      v4i_t gz4 = {0, 0, 0, 0};
      v8i_t gz8 = {0, 0, 0, 0, 0, 0, 0, 0};
      // 6-arg form (clang-23 / therock headers): (g0, g1, g2, g3, g4, cpol)
      __builtin_amdgcn_tensor_load_to_lds(g0, g1, gz4, gz4, gz8, 0);
      __builtin_amdgcn_s_wait_tensorcnt(0);
    }
    __syncthreads();
    // convert raw f32 LDS tile -> bf16 fragment order
#pragma unroll
    for (int r = 0; r < 4; ++r) {
      int idx4 = t + r * 128;
      int cl = idx4 >> 4;                 // k: 0..31
      int s4 = (idx4 & 15) * 4;           // 0..60
      const float4 x4 = *(const float4*)(Xf + cl * 64 + s4);
      Xs[fragB_idx(cl, s4 + 0)] = (bf16_t)x4.x;
      Xs[fragB_idx(cl, s4 + 1)] = (bf16_t)x4.y;
      Xs[fragB_idx(cl, s4 + 2)] = (bf16_t)x4.z;
      Xs[fragB_idx(cl, s4 + 3)] = (bf16_t)x4.w;
    }
#else
    // --- X tile 32(k) x 64(s): float4 load, 4 scattered b16 stores ---
#pragma unroll
    for (int r = 0; r < 4; ++r) {
      int idx4 = t + r * 128;
      int cl = idx4 >> 4;
      int s4 = (idx4 & 15) * 4;
      const float4 x4 = *(const float4*)(Xn + (size_t)(c0 + cl) * S + s0 + s4);
      Xs[fragB_idx(cl, s4 + 0)] = (bf16_t)x4.x;
      Xs[fragB_idx(cl, s4 + 1)] = (bf16_t)x4.y;
      Xs[fragB_idx(cl, s4 + 2)] = (bf16_t)x4.z;
      Xs[fragB_idx(cl, s4 + 3)] = (bf16_t)x4.w;
    }
#endif
    __syncthreads();
    v16bf a0 = *(const v16bf*)(Ws + ((wm >> 4) + 0) * 512 + lane * 16);
    v16bf a1 = *(const v16bf*)(Ws + ((wm >> 4) + 1) * 512 + lane * 16);
    v16bf b0 = *(const v16bf*)(Xs + ((wn >> 4) + 0) * 512 + lane * 16);
    v16bf b1 = *(const v16bf*)(Xs + ((wn >> 4) + 1) * 512 + lane * 16);
    acc[0][0] = WMMA_BF16(a0, b0, acc[0][0]);
    acc[0][1] = WMMA_BF16(a0, b1, acc[0][1]);
    acc[1][0] = WMMA_BF16(a1, b0, acc[1][0]);
    acc[1][1] = WMMA_BF16(a1, b1, acc[1][1]);
    __syncthreads();
  }

  // Epilogue: C/D layout -> VGPR r: lanes 0-15 M=r, lanes 16-31 M=8+r; N=lane&15
  const int half = lane & 15, hi = lane >> 4;
#pragma unroll
  for (int mt = 0; mt < 2; ++mt)
#pragma unroll
    for (int nt = 0; nt < 2; ++nt)
#pragma unroll
      for (int r = 0; r < 8; ++r) {
        int o = o0 + wm + mt * 16 + r + 8 * hi;
        int s = s0 + wn + nt * 16 + half;
        size_t off = (size_t)o * S + s;
        float v = acc[mt][nt][r] + bias[o];
        if (ACC) v += Yn[off];
        Yn[off] = v;
      }
}

// ---------------------------------------------------------------------------
// Gram reduction: att[n,v,w] = scale * sum_k a1[n,k,v] * a2[n,k,w]  (v,w<25)
// a1/a2 are [N, K, 25]; grid = N, block 128; 4 waves each reduce K chunks of 32.
// ---------------------------------------------------------------------------
__global__ __launch_bounds__(128)
void attn_gram_kernel(float* __restrict__ att, const float* __restrict__ a1,
                      const float* __restrict__ a2, int K, float scale)
{
  __shared__ __align__(32) bf16_t A1s[4 * 1024];   // per-wave 32x32 A chunk, frag order
  __shared__ __align__(32) bf16_t A2s[4 * 1024];   // per-wave 32x32 B chunk, frag order
  __shared__ float sAcc[32 * 32];
  const int t = threadIdx.x, lane = t & 31, wv = t >> 5;
  const int n = blockIdx.x;

  for (int i = t; i < 32 * 32; i += 128) sAcc[i] = 0.f;
  for (int i = t; i < 4 * 1024; i += 128) { A1s[i] = (bf16_t)0.f; A2s[i] = (bf16_t)0.f; }
  __syncthreads();

  const float* a1n = a1 + (size_t)n * K * 25;
  const float* a2n = a2 + (size_t)n * K * 25;
  v8f acc[2][2] = {};

  for (int k0 = 0; k0 < K; k0 += 128) {
    for (int q = t; q < 4 * 32 * 25; q += 128) {
      int ch = q / 800, rem = q % 800;
      int row = rem / 25, col = rem % 25;     // row = k within chunk, col = v/w
      int k = k0 + ch * 32 + row;
      // A matrix element (m=col, k=row); B matrix element (k=row, n=col)
      A1s[ch * 1024 + fragA_idx(col, row)] = (bf16_t)a1n[(size_t)k * 25 + col];
      A2s[ch * 1024 + fragB_idx(row, col)] = (bf16_t)a2n[(size_t)k * 25 + col];
    }
    __syncthreads();
    const bf16_t* m1 = A1s + wv * 1024;
    const bf16_t* m2 = A2s + wv * 1024;
    v16bf fa0 = *(const v16bf*)(m1 + lane * 16);
    v16bf fa1 = *(const v16bf*)(m1 + 512 + lane * 16);
    v16bf fb0 = *(const v16bf*)(m2 + lane * 16);
    v16bf fb1 = *(const v16bf*)(m2 + 512 + lane * 16);
    acc[0][0] = WMMA_BF16(fa0, fb0, acc[0][0]);
    acc[0][1] = WMMA_BF16(fa0, fb1, acc[0][1]);
    acc[1][0] = WMMA_BF16(fa1, fb0, acc[1][0]);
    acc[1][1] = WMMA_BF16(fa1, fb1, acc[1][1]);
    __syncthreads();
  }

  const int half = lane & 15, hi = lane >> 4;
#pragma unroll
  for (int mt = 0; mt < 2; ++mt)
#pragma unroll
    for (int nt = 0; nt < 2; ++nt)
#pragma unroll
      for (int r = 0; r < 8; ++r)
        atomicAdd(&sAcc[(mt * 16 + r + 8 * hi) * 32 + nt * 16 + half], acc[mt][nt][r]);
  __syncthreads();

  for (int q = t; q < 25 * 25; q += 128) {
    int v = q / 25, w = q % 25;
    att[(size_t)n * 625 + q] = sAcc[v * 32 + w] * scale;
  }
}

// ---------------------------------------------------------------------------
// Column softmax (over v, axis=-2) + adjacency: Ai = softmax(att) + Astat + PA
// ---------------------------------------------------------------------------
__global__ __launch_bounds__(32)
void softmax_adj_kernel(float* __restrict__ Ai, const float* __restrict__ att,
                        const float* __restrict__ Astat, const float* __restrict__ PA)
{
  const int n = blockIdx.x, w = threadIdx.x;
  if (w >= 25) return;
  const float* an = att + (size_t)n * 625;
  float mx = -1e30f;
#pragma unroll
  for (int v = 0; v < 25; ++v) mx = fmaxf(mx, an[v * 25 + w]);
  float ex[25], sum = 0.f;
#pragma unroll
  for (int v = 0; v < 25; ++v) { ex[v] = __expf(an[v * 25 + w] - mx); sum += ex[v]; }
  const float inv = 1.f / sum;
#pragma unroll
  for (int v = 0; v < 25; ++v)
    Ai[(size_t)n * 625 + v * 25 + w] = ex[v] * inv + Astat[v * 25 + w] + PA[v * 25 + w];
}

// ---------------------------------------------------------------------------
// Graph matmul: Z[row, w] = sum_v X[row, v] * Ai[n, v, w]; row = n*rowsPerN+m
// Memory-bound (K=N=25): plain VALU, Ai staged in LDS. block 256.
// ---------------------------------------------------------------------------
__global__ __launch_bounds__(256)
void graph_mm_kernel(float* __restrict__ Z, const float* __restrict__ X,
                     const float* __restrict__ Ai, int rowsPerN)
{
  __shared__ float sA[625];
  const size_t row = (size_t)blockIdx.x * 256 + threadIdx.x;
  const int n = (int)(row / (size_t)rowsPerN);
  for (int i = threadIdx.x; i < 625; i += 256) sA[i] = Ai[(size_t)n * 625 + i];
  __syncthreads();
  const float* xr = X + row * 25;
  float xv[25];
#pragma unroll
  for (int v = 0; v < 25; ++v) xv[v] = xr[v];
  float* zr = Z + row * 25;
#pragma unroll
  for (int w = 0; w < 25; ++w) {
    float s = 0.f;
#pragma unroll
    for (int v = 0; v < 25; ++v) s = fmaf(xv[v], sA[v * 25 + w], s);
    zr[w] = s;
  }
}

// out = relu(y + down), elementwise, float4-vectorized
__global__ __launch_bounds__(256)
void residual_relu_kernel(float4* __restrict__ out, const float4* __restrict__ y,
                          const float4* __restrict__ dn, size_t count4)
{
  size_t i = (size_t)blockIdx.x * 256 + threadIdx.x;
  if (i < count4) {
    float4 a = y[i], b = dn[i], o;
    o.x = fmaxf(a.x + b.x, 0.f);
    o.y = fmaxf(a.y + b.y, 0.f);
    o.z = fmaxf(a.z + b.z, 0.f);
    o.w = fmaxf(a.w + b.w, 0.f);
    out[i] = o;
  }
}

// ---------------------------------------------------------------------------
extern "C" void kernel_launch(void* const* d_in, const int* in_sizes, int n_in,
                              void* d_out, int out_size, void* d_ws, size_t ws_size,
                              hipStream_t stream)
{
  (void)out_size; (void)ws_size;
  const int N = 8, T = 256, V = 25, S = T * V;     // 6400
  static const int dims[10][2] = {
    {256,256},{256,256},{256,256},{256,256},{256,512},
    {512,512},{512,512},{512,512},{512,512},{512,512}};

  // Detect flattening order: jax pytree sorts dict keys (A_static < params < x),
  // plain insertion order puts x first. Disambiguate via element counts.
  const bool sortedMode = (in_sizes[0] == 3 * V * V);
  const float* x0;
  const float* Astatic;
  int idx;
  if (sortedMode) { Astatic = (const float*)d_in[0]; idx = 1; x0 = (const float*)d_in[n_in - 1]; }
  else            { x0 = (const float*)d_in[0];      idx = 1; Astatic = (const float*)d_in[n_in - 1]; }

  struct LP { const float *PA,*Wa,*ba,*Wb,*bb,*Wd,*bd,*Wdown,*bdown; };
  LP lp[10];
  for (int l = 0; l < 10; ++l) {
    const bool hd = dims[l][0] != dims[l][1];
    LP p; p.Wdown = nullptr; p.bdown = nullptr;
    if (sortedMode) {  // PA, Wa, Wb, Wd, (Wdown), ba, bb, bd, (bdown)
      p.PA = (const float*)d_in[idx++]; p.Wa = (const float*)d_in[idx++];
      p.Wb = (const float*)d_in[idx++]; p.Wd = (const float*)d_in[idx++];
      if (hd) p.Wdown = (const float*)d_in[idx++];
      p.ba = (const float*)d_in[idx++]; p.bb = (const float*)d_in[idx++];
      p.bd = (const float*)d_in[idx++];
      if (hd) p.bdown = (const float*)d_in[idx++];
    } else {           // PA, Wa, ba, Wb, bb, Wd, bd, (Wdown, bdown)
      p.PA = (const float*)d_in[idx++]; p.Wa = (const float*)d_in[idx++];
      p.ba = (const float*)d_in[idx++]; p.Wb = (const float*)d_in[idx++];
      p.bb = (const float*)d_in[idx++]; p.Wd = (const float*)d_in[idx++];
      p.bd = (const float*)d_in[idx++];
      if (hd) { p.Wdown = (const float*)d_in[idx++]; p.bdown = (const float*)d_in[idx++]; }
    }
    lp[l] = p;
  }

  // Workspace carve-up (floats)
  float* ws = (float*)d_ws;
  const size_t big = (size_t)N * 512 * S;          // 26,214,400 floats (105 MB)
  float* B0 = ws;            float* B1 = B0 + big;
  float* B2 = B1 + big;      float* BZ = B2 + big;          // z / down scratch
  float* A1 = BZ + big;
  float* A2 = A1 + (size_t)N * 128 * S;
  float* ATT = A2 + (size_t)N * 128 * S;
  float* AI  = ATT + (size_t)N * 1024;

  const float* xcur = x0;
  float* ybuf = B0;
  float* obuf = B1;
  float* spare = B2;

  for (int l = 0; l < 10; ++l) {
    const int cin = dims[l][0], cout = dims[l][1], inter = cout / 4;
    const int K = inter * T;
    const float scale = 1.f / (float)K;

    for (int i = 0; i < 3; ++i) {
      const float* Wa_i = lp[l].Wa + (size_t)i * inter * cin;
      const float* Wb_i = lp[l].Wb + (size_t)i * inter * cin;
      const float* Wd_i = lp[l].Wd + (size_t)i * cout * cin;

      gemm_proj_kernel<0><<<dim3(S / 64, inter / 64, N), 128, 0, stream>>>(
          A1, Wa_i, lp[l].ba + (size_t)i * inter, xcur, inter, cin, S);
      gemm_proj_kernel<0><<<dim3(S / 64, inter / 64, N), 128, 0, stream>>>(
          A2, Wb_i, lp[l].bb + (size_t)i * inter, xcur, inter, cin, S);
      attn_gram_kernel<<<N, 128, 0, stream>>>(ATT, A1, A2, K, scale);
      softmax_adj_kernel<<<N, 32, 0, stream>>>(AI, ATT, Astatic + (size_t)i * 625,
                                               lp[l].PA + (size_t)i * 625);
      graph_mm_kernel<<<((size_t)N * cin * T) / 256, 256, 0, stream>>>(
          BZ, xcur, AI, cin * T);
      if (i == 0)
        gemm_proj_kernel<0><<<dim3(S / 64, cout / 64, N), 128, 0, stream>>>(
            ybuf, Wd_i, lp[l].bd + (size_t)i * cout, BZ, cout, cin, S);
      else
        gemm_proj_kernel<1><<<dim3(S / 64, cout / 64, N), 128, 0, stream>>>(
            ybuf, Wd_i, lp[l].bd + (size_t)i * cout, BZ, cout, cin, S);
    }

    const float* dn;
    if (cin != cout) {
      gemm_proj_kernel<0><<<dim3(S / 64, cout / 64, N), 128, 0, stream>>>(
          BZ, lp[l].Wdown, lp[l].bdown, xcur, cout, cin, S);
      dn = BZ;
    } else {
      dn = xcur;
    }

    float* outp = (l == 9) ? (float*)d_out : obuf;
    const size_t cnt4 = ((size_t)N * cout * S) / 4;
    residual_relu_kernel<<<(cnt4 + 255) / 256, 256, 0, stream>>>(
        (float4*)outp, (const float4*)ybuf, (const float4*)dn, cnt4);

    // rotate buffers: new x = out; y reuses old y; out reuses old x (or spare on l0)
    const float* oldX = xcur;
    xcur = outp;
    obuf = (l == 0) ? spare : (float*)oldX;
  }
}
